// TransformerLayer_33337536151564
// MI455X (gfx1250) — matrix operand order repaired and checked by reading, not compiled, not run on hardware
//
#include <hip/hip_runtime.h>

typedef __bf16 bf16_t;
typedef __attribute__((ext_vector_type(16))) __bf16 v16bf;
typedef __attribute__((ext_vector_type(8)))  __bf16 v8bf;
typedef __attribute__((ext_vector_type(8)))  float  v8f;

#define HID   1024
#define FFD   4096
#define SEQ   4096
#define BT    4
#define MTOT  (BT*SEQ)     /* 16384 rows of the flattened activation matrix */
#define NHEADS 16
#define HDIM  64
#define LN_EPS 1e-12f

enum { EP_F32_BIAS = 0, EP_VT = 1, EP_RES = 2, EP_GELU = 3, EP_KVT = 4, EP_CTX = 5 };

// ---------------------------------------------------------------- conversions
__global__ void cvt_f32_to_bf16(const float* __restrict__ in, bf16_t* __restrict__ out,
                                long long n) {
  long long i = (long long)blockIdx.x * blockDim.x + threadIdx.x;
  long long stride = (long long)gridDim.x * blockDim.x;
  for (; i < n; i += stride) out[i] = (bf16_t)in[i];
}

// W[K][N] f32 (row-major, fan_in x fan_out) -> Wt[N][K] bf16
__global__ void cvt_transpose_to_bf16(const float* __restrict__ in, bf16_t* __restrict__ out,
                                      int K, int N) {
  long long total = (long long)K * N;
  long long i = (long long)blockIdx.x * blockDim.x + threadIdx.x;
  long long stride = (long long)gridDim.x * blockDim.x;
  for (; i < total; i += stride) {
    int k = (int)(i / N);
    int n = (int)(i % N);
    out[(long long)n * K + k] = (bf16_t)in[i];
  }
}

// ---------------------------------------------------------------- generic WMMA GEMM
// C[m][n] = sum_k A[aOff + m*lda + k] * Bt[bOff + n*ldb + k]   (Bt is [N][K])
// BM=128, BN=128, BK=32; 256 threads = 8 waves (4 M x 2 N); wave tile 32x64 = 2x4 WMMA.
// Double-buffered LDS: one barrier per K-step.
__global__ void __launch_bounds__(256) gemm_bf16_wmma(
    const bf16_t* __restrict__ A, long long lda, long long aHi, long long aLo,
    const bf16_t* __restrict__ B, long long ldb, long long bHi, long long bLo,
    const float* __restrict__ bias,
    int M, int N, int K,
    int mode, float scale,
    float* __restrict__ outF, bf16_t* __restrict__ outB,
    const float* __restrict__ res,
    long long ldo, long long oHi, long long oLo)
{
  __shared__ __attribute__((aligned(32))) bf16_t ldsA[2][128 * 32];
  __shared__ __attribute__((aligned(32))) bf16_t ldsB[2][128 * 32];

  const int t    = threadIdx.x;
  const int w    = t >> 5;
  const int l    = t & 31;
  const int half = l >> 4;        // lane group: 0 -> lanes 0-15, 1 -> lanes 16-31
  const int lq   = l & 15;
  const int wm   = w & 3;         // wave M block (4 x 32 = 128)
  const int wn   = w >> 2;        // wave N block (2 x 64 = 128)

  const int z = blockIdx.z;       // batch = (b*16+h) for attention GEMMs
  const long long aOff = (long long)(z >> 4) * aHi + (long long)(z & 15) * aLo;
  const long long bOff = (long long)(z >> 4) * bHi + (long long)(z & 15) * bLo;
  const long long oOff = (long long)(z >> 4) * oHi + (long long)(z & 15) * oLo;

  const int mBase = blockIdx.y * 128;
  const int nBase = blockIdx.x * 128;

  // Staging: each 128x32 bf16 tile = 512 16B chunks; 2 per thread (rows r, r+64).
  const int row0 = t >> 2;             // 0..63
  const int ko   = (t & 3) * 8;        // 0,8,16,24
  const int sOf0 = row0 * 32 + ko;
  const int sOf1 = sOf0 + 64 * 32;

  int gra0 = mBase + row0;       if (gra0 > M - 1) gra0 = M - 1;
  int gra1 = mBase + row0 + 64;  if (gra1 > M - 1) gra1 = M - 1;
  int grb0 = nBase + row0;       if (grb0 > N - 1) grb0 = N - 1;
  int grb1 = nBase + row0 + 64;  if (grb1 > N - 1) grb1 = N - 1;

  const bf16_t* pA0 = A + aOff + (long long)gra0 * lda + ko;
  const bf16_t* pA1 = A + aOff + (long long)gra1 * lda + ko;
  const bf16_t* pB0 = B + bOff + (long long)grb0 * ldb + ko;
  const bf16_t* pB1 = B + bOff + (long long)grb1 * ldb + ko;

  const int nIter = K >> 5;

  // prologue: tile 0 -> buffer 0
  {
    uint4 a0 = *(const uint4*)pA0;  pA0 += 32;
    uint4 a1 = *(const uint4*)pA1;  pA1 += 32;
    uint4 b0 = *(const uint4*)pB0;  pB0 += 32;
    uint4 b1 = *(const uint4*)pB1;  pB1 += 32;
    *(uint4*)&ldsA[0][sOf0] = a0;
    *(uint4*)&ldsA[0][sOf1] = a1;
    *(uint4*)&ldsB[0][sOf0] = b0;
    *(uint4*)&ldsB[0][sOf1] = b1;
  }
  __syncthreads();

  v8f acc[2][4] = {};

  for (int it = 0; it < nIter; ++it) {
    const int cur = it & 1;
    const int nxt = cur ^ 1;

    // prefetch next tile into registers (hidden under the WMMAs below)
    uint4 a0, a1, b0, b1;
    const bool more = (it + 1) < nIter;
    if (more) {
      a0 = *(const uint4*)pA0;  pA0 += 32;
      a1 = *(const uint4*)pA1;  pA1 += 32;
      b0 = *(const uint4*)pB0;  pB0 += 32;
      b1 = *(const uint4*)pB1;  pB1 += 32;
    }

    // A fragment: lane = row M=lq, K = {0..7,16..23} + half*8 -> two 16B LDS reads
    v16bf afrag[2];
#pragma unroll
    for (int mi = 0; mi < 2; ++mi) {
      const bf16_t* pa = &ldsA[cur][(wm * 32 + mi * 16 + lq) * 32];
      v8bf alo = *(const v8bf*)(pa + half * 8);
      v8bf ahi = *(const v8bf*)(pa + 16 + half * 8);
#pragma unroll
      for (int j = 0; j < 8; ++j) { afrag[mi][j] = alo[j]; afrag[mi][j + 8] = ahi[j]; }
    }
    // B fragment: lane = col N=lq, K = {0..7,16..23} + half*8 -> two 16B LDS reads from [N][K] tile
    v16bf bfrag[4];
#pragma unroll
    for (int ni = 0; ni < 4; ++ni) {
      const bf16_t* pb = &ldsB[cur][(wn * 64 + ni * 16 + lq) * 32];
      v8bf blo = *(const v8bf*)(pb + half * 8);
      v8bf bhi = *(const v8bf*)(pb + 16 + half * 8);
#pragma unroll
      for (int j = 0; j < 8; ++j) { bfrag[ni][j] = blo[j]; bfrag[ni][j + 8] = bhi[j]; }
    }

#pragma unroll
    for (int mi = 0; mi < 2; ++mi)
#pragma unroll
      for (int ni = 0; ni < 4; ++ni)
        acc[mi][ni] = __builtin_amdgcn_wmma_f32_16x16x32_bf16(
            false, afrag[mi], false, bfrag[ni], (short)0, acc[mi][ni], false, false);

    if (more) {
      *(uint4*)&ldsA[nxt][sOf0] = a0;
      *(uint4*)&ldsA[nxt][sOf1] = a1;
      *(uint4*)&ldsB[nxt][sOf0] = b0;
      *(uint4*)&ldsB[nxt][sOf1] = b1;
    }
    __syncthreads();
  }

  // epilogue: C lane l, reg r -> M = r + 8*half, N = lq (within each 16x16 tile)
#pragma unroll
  for (int mi = 0; mi < 2; ++mi) {
#pragma unroll
    for (int ni = 0; ni < 4; ++ni) {
#pragma unroll
      for (int r = 0; r < 8; ++r) {
        int m = mBase + wm * 32 + mi * 16 + half * 8 + r;
        int n = nBase + wn * 64 + ni * 16 + lq;
        if (m >= M || n >= N) continue;
        float c = acc[mi][ni][r];
        switch (mode) {
          case EP_F32_BIAS:
            outF[oOff + (long long)m * ldo + n] = c + bias[n];
            break;
          case EP_VT: {  // v projection -> vt[b][h][e][s] bf16
            float vv = c + bias[n];
            long long idx = ((((long long)(m >> 12)) * 16 + (n >> 6)) * 64 + (n & 63)) * 4096LL
                            + (m & 4095);
            outB[idx] = (bf16_t)vv;
            break; }
          case EP_RES:
            outF[oOff + (long long)m * ldo + n] =
                c + bias[n] + res[oOff + (long long)m * ldo + n];
            break;
          case EP_GELU: {
            float xx = c + bias[n];
            float g  = 0.5f * xx * (1.0f + erff(xx * 0.70710678118f));
            outB[oOff + (long long)m * ldo + n] = (bf16_t)g;
            break; }
          case EP_KVT:  // store C^T: kvt[e][d] = kv[d][e]
            outB[oOff + (long long)n * 64 + m] = (bf16_t)c;
            break;
          case EP_CTX:
            outB[oOff + (long long)m * ldo + n] = (bf16_t)(c * scale);
            break;
        }
      }
    }
  }
}

// ---------------------------------------------------------------- ELU + L2 feature map
// qf/kf: f32 [MTOT][HID]. Writes qn bf16 row-major and kn transposed [b][h][d][s].
__global__ void featmap_kernel(const float* __restrict__ qf, const float* __restrict__ kf,
                               bf16_t* __restrict__ qnb, bf16_t* __restrict__ knt)
{
  int idx = blockIdx.x * blockDim.x + threadIdx.x;
  if (idx >= MTOT * NHEADS) return;
  int m = idx >> 4, h = idx & 15;
  long long base = (long long)m * HID + h * HDIM;
  float sq = 0.f, sk = 0.f;
  for (int d = 0; d < HDIM; ++d) {
    float q = qf[base + d]; float e = q > 0.f ? q : (expf(q) - 1.f); sq += e * e;
    float k = kf[base + d]; float f = k > 0.f ? k : (expf(k) - 1.f); sk += f * f;
  }
  float rq = rsqrtf(sq), rk = rsqrtf(sk);
  int b = m >> 12, s = m & 4095;
  long long ktbase = (((long long)b * 16 + h) * 64) * 4096LL + s;
  for (int d = 0; d < HDIM; ++d) {
    float q = qf[base + d]; float e = q > 0.f ? q : (expf(q) - 1.f);
    qnb[base + d] = (bf16_t)(e * rq);
    float k = kf[base + d]; float f = k > 0.f ? k : (expf(k) - 1.f);
    knt[ktbase + (long long)d * 4096] = (bf16_t)(f * rk);
  }
}

// ---------------------------------------------------------------- row LayerNorm (H=1024)
__global__ void __launch_bounds__(256) layernorm_kernel(
    const float* __restrict__ in, const float* __restrict__ g, const float* __restrict__ b,
    float* __restrict__ outF, bf16_t* __restrict__ outB)
{
  __shared__ float red[256];
  const int t = threadIdx.x;
  const long long base = (long long)blockIdx.x * HID;
  float x0 = in[base + t + 0];
  float x1 = in[base + t + 256];
  float x2 = in[base + t + 512];
  float x3 = in[base + t + 768];
  red[t] = x0 + x1 + x2 + x3;
  __syncthreads();
  for (int o = 128; o > 0; o >>= 1) { if (t < o) red[t] += red[t + o]; __syncthreads(); }
  float mu = red[0] * (1.0f / HID);
  __syncthreads();
  float d0 = x0 - mu, d1 = x1 - mu, d2 = x2 - mu, d3 = x3 - mu;
  red[t] = d0 * d0 + d1 * d1 + d2 * d2 + d3 * d3;
  __syncthreads();
  for (int o = 128; o > 0; o >>= 1) { if (t < o) red[t] += red[t + o]; __syncthreads(); }
  float rstd = rsqrtf(red[0] * (1.0f / HID) + LN_EPS);
  float xs[4] = { x0, x1, x2, x3 };
#pragma unroll
  for (int i = 0; i < 4; ++i) {
    int n = t + i * 256;
    float y = (xs[i] - mu) * rstd * g[n] + b[n];
    outF[base + n] = y;
    outB[base + n] = (bf16_t)y;
  }
}

// ---------------------------------------------------------------- launcher
extern "C" void kernel_launch(void* const* d_in, const int* in_sizes, int n_in,
                              void* d_out, int out_size, void* d_ws, size_t ws_size,
                              hipStream_t stream)
{
  const float* x   = (const float*)d_in[0];
  const float* Wq  = (const float*)d_in[1];  const float* bq  = (const float*)d_in[2];
  const float* Wk  = (const float*)d_in[3];  const float* bk  = (const float*)d_in[4];
  const float* Wv  = (const float*)d_in[5];  const float* bv  = (const float*)d_in[6];
  const float* Wo  = (const float*)d_in[7];  const float* bo  = (const float*)d_in[8];
  const float* g1  = (const float*)d_in[9];  const float* be1 = (const float*)d_in[10];
  const float* W1  = (const float*)d_in[11]; const float* b1  = (const float*)d_in[12];
  const float* W2  = (const float*)d_in[13]; const float* b2  = (const float*)d_in[14];
  const float* g2  = (const float*)d_in[15]; const float* be2 = (const float*)d_in[16];

  char* wsp = (char*)d_ws;
  size_t off = 0;
  auto alloc = [&](size_t bytes) -> char* {
    char* p = wsp + off;
    off = (off + bytes + 255) & ~(size_t)255;
    return p;
  };

  bf16_t* xb   = (bf16_t*)alloc((size_t)MTOT * HID * 2);
  bf16_t* wqt  = (bf16_t*)alloc((size_t)HID * HID * 2);
  bf16_t* wkt  = (bf16_t*)alloc((size_t)HID * HID * 2);
  bf16_t* wvt  = (bf16_t*)alloc((size_t)HID * HID * 2);
  bf16_t* wot  = (bf16_t*)alloc((size_t)HID * HID * 2);
  bf16_t* w1t  = (bf16_t*)alloc((size_t)HID * FFD * 2);
  bf16_t* w2t  = (bf16_t*)alloc((size_t)FFD * HID * 2);
  float*  qf   = (float*) alloc((size_t)MTOT * HID * 4);
  float*  kf   = (float*) alloc((size_t)MTOT * HID * 4);
  bf16_t* qnb  = (bf16_t*)alloc((size_t)MTOT * HID * 2);
  bf16_t* knt  = (bf16_t*)alloc((size_t)MTOT * HID * 2);
  bf16_t* vt   = (bf16_t*)alloc((size_t)MTOT * HID * 2);
  bf16_t* kvt  = (bf16_t*)alloc((size_t)64 * 64 * 64 * 2);
  bf16_t* ctxb = (bf16_t*)alloc((size_t)MTOT * HID * 2);
  float*  attnF= (float*) alloc((size_t)MTOT * HID * 4);
  bf16_t* hb   = (bf16_t*)alloc((size_t)MTOT * FFD * 2);
  // buffer reuse (lifetimes don't overlap)
  float*  attnPre = qf;   // pre-LN attention sum (qf dead after featmap)
  float*  outPre  = kf;   // pre-LN final sum     (kf dead after featmap)
  bf16_t* attnb   = xb;   // bf16 LN1 output      (xb dead after V projection)
  bf16_t* outb    = qnb;  // bf16 LN2 output      (qnb dead after ctx GEMM)

  dim3 blk(256);

  // 1) precision conversion / weight transpose
  cvt_f32_to_bf16<<<dim3(2048), blk, 0, stream>>>(x, xb, (long long)MTOT * HID);
  cvt_transpose_to_bf16<<<dim3(1024), blk, 0, stream>>>(Wq, wqt, HID, HID);
  cvt_transpose_to_bf16<<<dim3(1024), blk, 0, stream>>>(Wk, wkt, HID, HID);
  cvt_transpose_to_bf16<<<dim3(1024), blk, 0, stream>>>(Wv, wvt, HID, HID);
  cvt_transpose_to_bf16<<<dim3(1024), blk, 0, stream>>>(Wo, wot, HID, HID);
  cvt_transpose_to_bf16<<<dim3(4096), blk, 0, stream>>>(W1, w1t, HID, FFD);
  cvt_transpose_to_bf16<<<dim3(4096), blk, 0, stream>>>(W2, w2t, FFD, HID);

  // 2) QKV projections (M=16384, N=1024, K=1024)
  dim3 gProj(HID / 128, MTOT / 128, 1);
  gemm_bf16_wmma<<<gProj, blk, 0, stream>>>(xb, HID, 0, 0, wqt, HID, 0, 0, bq,
      MTOT, HID, HID, EP_F32_BIAS, 1.f, qf, nullptr, nullptr, HID, 0, 0);
  gemm_bf16_wmma<<<gProj, blk, 0, stream>>>(xb, HID, 0, 0, wkt, HID, 0, 0, bk,
      MTOT, HID, HID, EP_F32_BIAS, 1.f, kf, nullptr, nullptr, HID, 0, 0);
  gemm_bf16_wmma<<<gProj, blk, 0, stream>>>(xb, HID, 0, 0, wvt, HID, 0, 0, bv,
      MTOT, HID, HID, EP_VT, 1.f, nullptr, vt, nullptr, 0, 0, 0);

  // 3) ELU feature map + L2 normalize
  featmap_kernel<<<dim3((MTOT * NHEADS + 255) / 256), blk, 0, stream>>>(qf, kf, qnb, knt);

  // 4) kv = kn^T @ v per (b,h): 64 batched 64x64x4096 GEMMs -> kvt[e][d]
  gemm_bf16_wmma<<<dim3(1, 1, 64), blk, 0, stream>>>(
      knt, 4096, 16LL * 64 * 4096, 64LL * 4096,
      vt,  4096, 16LL * 64 * 4096, 64LL * 4096,
      nullptr, 64, 64, 4096, EP_KVT, 1.f,
      nullptr, kvt, nullptr, 0, 16LL * 4096, 4096);

  // 5) ctx = (qn @ kv) / sqrt(HD) per (b,h): 64 batched 4096x64x64 GEMMs
  gemm_bf16_wmma<<<dim3(1, SEQ / 128, 64), blk, 0, stream>>>(
      qnb, HID, (long long)SEQ * HID, 64,
      kvt, 64,  16LL * 4096, 4096,
      nullptr, SEQ, 64, 64, EP_CTX, 0.125f,
      nullptr, ctxb, nullptr, HID, (long long)SEQ * HID, 64);

  // 6) attn_out pre-LN = ctx @ Wo + bo + x
  gemm_bf16_wmma<<<gProj, blk, 0, stream>>>(ctxb, HID, 0, 0, wot, HID, 0, 0, bo,
      MTOT, HID, HID, EP_RES, 1.f, attnPre, nullptr, x, HID, 0, 0);
  layernorm_kernel<<<dim3(MTOT), blk, 0, stream>>>(attnPre, g1, be1, attnF, attnb);

  // 7) feed-forward
  gemm_bf16_wmma<<<dim3(FFD / 128, MTOT / 128, 1), blk, 0, stream>>>(
      attnb, HID, 0, 0, w1t, HID, 0, 0, b1,
      MTOT, FFD, HID, EP_GELU, 1.f, nullptr, hb, nullptr, FFD, 0, 0);
  gemm_bf16_wmma<<<gProj, blk, 0, stream>>>(hb, FFD, 0, 0, w2t, FFD, 0, 0, b2,
      MTOT, HID, FFD, EP_RES, 1.f, outPre, nullptr, attnF, HID, 0, 0);
  layernorm_kernel<<<dim3(MTOT), blk, 0, stream>>>(outPre, g2, be2, (float*)d_out, outb);

  (void)in_sizes; (void)n_in; (void)out_size; (void)ws_size;
}
